// RNN_fish_2095944040501
// MI455X (gfx1250) — matrix operand order repaired and checked by reading, not compiled, not run on hardware
//
#include <hip/hip_runtime.h>

#define B_  64
#define S_  512
#define I_  128
#define H_  2048
#define O_  128
#define NBLK 32
#define ALPHA_F 0.2f
#define NOISE_F 0.005f

typedef __bf16 v16bf __attribute__((ext_vector_type(16)));
typedef float  v8f   __attribute__((ext_vector_type(8)));
typedef unsigned int u32x4 __attribute__((ext_vector_type(4)));

// explicit global-address-space pointer types (force global_load/store, never flat)
typedef const __attribute__((address_space(1))) unsigned short* gcu16p;
typedef       __attribute__((address_space(1))) unsigned short* gu16p;
typedef const __attribute__((address_space(1))) float*          gcf32p;
typedef       __attribute__((address_space(1))) float*          gf32p;
typedef const __attribute__((address_space(1))) u32x4*          gcq16p;

// float -> bf16 bits, round-to-nearest-even (inputs are Gaussian; no NaN path needed)
static __device__ __forceinline__ unsigned short f2bf(float f) {
  unsigned u = __float_as_uint(f);
  unsigned r = u + 0x7FFFu + ((u >> 16) & 1u);
  return (unsigned short)(r >> 16);
}

// ---------------- one-time precompute kernels ----------------
// A[n*H + k] = bf16( ALPHA * g[k] * wrec[n][k] )  (same flat layout as wrec)
__global__ void prep_wrec(const float* __restrict__ wrec, const float* __restrict__ g,
                          unsigned short* __restrict__ Abf) {
  int idx = blockIdx.x * 256 + threadIdx.x;     // H*H threads
  int k = idx & (H_ - 1);
  Abf[idx] = f2bf(ALPHA_F * g[k] * wrec[idx]);
}
// WiT[n*I + k] = bf16( ALPHA * wi[k][n] )
__global__ void prep_wi(const float* __restrict__ wi, unsigned short* __restrict__ WiT) {
  int idx = blockIdx.x * 256 + threadIdx.x;     // H*I threads
  int n = idx >> 7;            // / I_
  int k = idx & (I_ - 1);
  WiT[idx] = f2bf(ALPHA_F * wi[k * H_ + n]);
}
// WoT[o*H + k] = bf16( wout[k][o] )
__global__ void prep_wout(const float* __restrict__ wout, unsigned short* __restrict__ WoT) {
  int idx = blockIdx.x * 256 + threadIdx.x;     // O*H threads
  int o = idx >> 11;           // / H_
  int k = idx & (H_ - 1);
  WoT[idx] = f2bf(wout[k * O_ + o]);
}
__global__ void prep_x(const float* __restrict__ x, unsigned short* __restrict__ Xbf) {
  int idx = blockIdx.x * 256 + threadIdx.x;     // B*S*I threads
  Xbf[idx] = f2bf(x[idx]);
}
__global__ void prep_h(const float* __restrict__ h0, unsigned short* __restrict__ hbuf0) {
  int idx = blockIdx.x * 256 + threadIdx.x;     // B*H threads
  hbuf0[idx] = f2bf(h0[idx & (H_ - 1)]);
}

// ---------------- fragment loaders (ISA 16-bit A / B layouts) ----------------
// A (16x32, M=li): two 16B chunks at k0 and k0+16, with k0 pre-offset by +8 for lanes>=16
static __device__ __forceinline__ v16bf loadA(gcu16p p) {
  union { v16bf v; u32x4 q[2]; } f;
  f.q[0] = *(gcq16p)(p);
  f.q[1] = *(gcq16p)(p + 16);
  return f.v;
}
// B (32x16, N=li): two 16B chunks at k0 and k0+16, with k0 pre-offset by +8 for lanes>=16
static __device__ __forceinline__ v16bf loadB(gcu16p p) {
  union { v16bf v; u32x4 q[2]; } f;
  f.q[0] = *(gcq16p)(p);
  f.q[1] = *(gcq16p)(p + 16);
  return f.v;
}
#define WMMA_BF16(a, b, c) \
  __builtin_amdgcn_wmma_f32_16x16x32_bf16(false, (a), false, (b), (short)0, (c), false, false)

// ---------------- persistent time-stepping kernel ----------------
__global__ void __launch_bounds__(256, 1) rnn_persistent(
    const unsigned short* Abf_,   // [H][H]  (N-major, K contiguous)
    const unsigned short* WiT_,   // [H][I]
    const unsigned short* WoT_,   // [O][H]
    const unsigned short* Xbf_,   // [B][S][I] bf16
    const float*          noise_, // [B][S][H] f32
    const float*          h0_,    // [H]
    unsigned short* hbuf0_, unsigned short* hbuf1_, // [B][H] bf16, double buffer
    float* out_,                  // [B][S][O] f32
    unsigned* barrier_ct)
{
  // force global address space on all bulk data
  gcu16p Abf   = (gcu16p)Abf_;
  gcu16p WiT   = (gcu16p)WiT_;
  gcu16p WoT   = (gcu16p)WoT_;
  gcu16p Xbf   = (gcu16p)Xbf_;
  gcf32p noise = (gcf32p)noise_;
  gcf32p h0    = (gcf32p)h0_;
  gu16p  hbuf0 = (gu16p)hbuf0_;
  gu16p  hbuf1 = (gu16p)hbuf1_;
  gf32p  out   = (gf32p)out_;

  const int lane = threadIdx.x & 31;
  const int wave = threadIdx.x >> 5;
  const int gw   = (int)blockIdx.x * 8 + wave;   // 0..255
  const int li   = lane & 15;
  const int hs8  = (lane >> 4) << 3;             // A-fragment K offset per lane half
  const int hs16 = (lane >> 4) << 4;             // 16 per lane half (unused)

  const int n0  = (gw & 127) * 16;               // 16 hidden columns owned by this wave
  const int m0  = (gw >> 7) * 32;                // 32 batch rows owned by this wave
  const int col = n0 + li;

  // fp32 hidden state lives in registers, in WMMA C/D tile layout (rows m0 and m0+16)
  v8f hA0, hA1;
  {
    float hv = h0[col];
    #pragma unroll
    for (int r = 0; r < 8; ++r) { hA0[r] = hv; hA1[r] = hv; }
  }

  const bool is_out_wave = ((gw & 7) == 7);      // 32 waves, one 16x16 out tile each
  const int oid = gw >> 3;
  const int om0 = (oid & 3) * 16;
  const int on0 = (oid >> 2) * 16;

  for (int t = 0; t < S_; ++t) {
    const gcu16p hcur = (t & 1) ? (gcu16p)hbuf1 : (gcu16p)hbuf0;
    const gu16p  hnxt = (t & 1) ? hbuf0 : hbuf1;

    // ---- out[:, t, :] = h_t @ wout (distributed 16x16 tiles) ----
    if (is_out_wave) {
      v8f oacc = {0.f, 0.f, 0.f, 0.f, 0.f, 0.f, 0.f, 0.f};
      gcu16p ap = hcur + (om0 + li) * H_ + hs8;
      gcu16p bp = WoT  + (on0 + li) * H_ + hs8;
      #pragma unroll 4
      for (int kt = 0; kt < H_ / 32; ++kt) {
        v16bf a = loadA(ap + kt * 32);
        v16bf b = loadB(bp + kt * 32);
        oacc = WMMA_BF16(a, b, oacc);
      }
      const int obr = om0 + hs8;
      const int oc  = on0 + li;
      #pragma unroll
      for (int r = 0; r < 8; ++r)
        out[((obr + r) * S_ + t) * O_ + oc] = oacc[r];
    }

    // ---- h_{t+1} = 0.8 h_t + 0.005 n_t + h_t@A + x_t@WiT (alpha folded into A/WiT) ----
    if (t < S_ - 1) {
      v8f acc0, acc1;
      #pragma unroll
      for (int r = 0; r < 8; ++r) {
        int b0 = m0 + hs8 + r;
        acc0[r] = 0.8f * hA0[r] + NOISE_F * noise[(b0        * S_ + t) * H_ + col];
        acc1[r] = 0.8f * hA1[r] + NOISE_F * noise[((b0 + 16) * S_ + t) * H_ + col];
      }
      {   // recurrent term, K = 2048 (weights streamed from L2)
        gcu16p a0p = hcur + (m0 + li) * H_ + hs8;
        gcu16p a1p = a0p + 16 * H_;
        gcu16p bp  = Abf + col * H_ + hs8;
        #pragma unroll 4
        for (int kt = 0; kt < H_ / 32; ++kt) {
          v16bf b  = loadB(bp  + kt * 32);
          v16bf a0 = loadA(a0p + kt * 32);
          v16bf a1 = loadA(a1p + kt * 32);
          acc0 = WMMA_BF16(a0, b, acc0);
          acc1 = WMMA_BF16(a1, b, acc1);
        }
      }
      {   // input term, K = 128
        gcu16p a0p = Xbf + ((m0 + li) * S_ + t) * I_ + hs8;
        gcu16p a1p = a0p + 16 * S_ * I_;
        gcu16p bp  = WiT + col * I_ + hs8;
        #pragma unroll
        for (int kt = 0; kt < I_ / 32; ++kt) {
          v16bf b  = loadB(bp  + kt * 32);
          v16bf a0 = loadA(a0p + kt * 32);
          v16bf a1 = loadA(a1p + kt * 32);
          acc0 = WMMA_BF16(a0, b, acc0);
          acc1 = WMMA_BF16(a1, b, acc1);
        }
      }
      hA0 = acc0; hA1 = acc1;
      #pragma unroll
      for (int r = 0; r < 8; ++r) {
        int b0 = m0 + hs8 + r;
        hnxt[b0        * H_ + col] = f2bf(acc0[r]);
        hnxt[(b0 + 16) * H_ + col] = f2bf(acc1[r]);
      }
    }

    // ---- device-wide barrier (monotonic counter, one arrival per block) ----
    __builtin_amdgcn_fence(__ATOMIC_RELEASE, "agent");
    __syncthreads();
    if (threadIdx.x == 0) {
      __hip_atomic_fetch_add(barrier_ct, 1u, __ATOMIC_RELEASE, __HIP_MEMORY_SCOPE_AGENT);
      const unsigned target = (unsigned)NBLK * (unsigned)(t + 1);
      while (__hip_atomic_load(barrier_ct, __ATOMIC_ACQUIRE, __HIP_MEMORY_SCOPE_AGENT) < target)
        __builtin_amdgcn_s_sleep(1);
    }
    __syncthreads();
    __builtin_amdgcn_fence(__ATOMIC_ACQUIRE, "agent");
  }
}

// ---------------- host launch ----------------
extern "C" void kernel_launch(void* const* d_in, const int* in_sizes, int n_in,
                              void* d_out, int out_size, void* d_ws, size_t ws_size,
                              hipStream_t stream) {
  const float* x     = (const float*)d_in[0];
  const float* noise = (const float*)d_in[1];
  const float* wi    = (const float*)d_in[2];
  const float* wrec  = (const float*)d_in[3];
  const float* wout  = (const float*)d_in[4];
  const float* g     = (const float*)d_in[5];
  const float* h0    = (const float*)d_in[6];
  float* out = (float*)d_out;

  // carve workspace (~18.4 MB total)
  char* w = (char*)d_ws;
  auto carve = [&](size_t bytes) -> char* {
    char* p = w; w += (bytes + 255) & ~(size_t)255; return p;
  };
  unsigned short* Abf   = (unsigned short*)carve((size_t)H_ * H_ * 2); // 8 MB
  unsigned short* WiT   = (unsigned short*)carve((size_t)H_ * I_ * 2); // 512 KB
  unsigned short* WoT   = (unsigned short*)carve((size_t)O_ * H_ * 2); // 512 KB
  unsigned short* Xbf   = (unsigned short*)carve((size_t)B_ * S_ * I_ * 2); // 8 MB
  unsigned short* hbuf0 = (unsigned short*)carve((size_t)B_ * H_ * 2); // 256 KB
  unsigned short* hbuf1 = (unsigned short*)carve((size_t)B_ * H_ * 2); // 256 KB
  unsigned*       ctr   = (unsigned*)carve(256);

  hipMemsetAsync(ctr, 0, 256, stream);

  prep_wrec<<<(H_ * H_) / 256, 256, 0, stream>>>(wrec, g, Abf);
  prep_wi  <<<(H_ * I_) / 256, 256, 0, stream>>>(wi, WiT);
  prep_wout<<<(O_ * H_) / 256, 256, 0, stream>>>(wout, WoT);
  prep_x   <<<(B_ * S_ * I_) / 256, 256, 0, stream>>>(x, Xbf);
  prep_h   <<<(B_ * H_) / 256, 256, 0, stream>>>(h0, hbuf0);

  rnn_persistent<<<NBLK, 256, 0, stream>>>(Abf, WiT, WoT, Xbf, noise, h0,
                                           hbuf0, hbuf1, out, ctr);
}
